// GemmaAttention_64871186039123
// MI455X (gfx1250) — matrix operand order repaired and checked by reading, not compiled, not run on hardware
//
#include <hip/hip_runtime.h>
#include <hip/hip_bf16.h>
#include <math.h>

// Problem constants (match reference)
#define S_LEN   2048
#define DMODEL  2048
#define HEADS   8
#define HDIM    256
#define BATCH   2
#define NEG_INF_V -1000000000.0f

typedef __attribute__((ext_vector_type(16))) __bf16 v16bf;
typedef __attribute__((ext_vector_type(8)))  float  v8f;

__device__ __forceinline__ int lane_id() { return (int)(threadIdx.x & 31); }

// ---------------------------------------------------------------------------
// WMMA bf16 fragment loaders, per CDNA5 ISA 7.12.2 layouts. Operands are
// pre-converted bf16 in memory -> pure b128 loads, no VALU cvt in the loop.
// A (16x32 MxK): lane<16: row M=lane, elems 0..7=K[k0..k0+7], 8..15=K[k0+16..23]
//                lane>=16: same rows, K[+8..15] / K[+24..31].
// B (32x16 KxN): lane n(0..15): col N=n, elems 0..7=K[k0..k0+7], 8..15=K[k0+16..23]; lanes 16-31: K[+8..15] / K[+24..31].
// ---------------------------------------------------------------------------
__device__ __forceinline__ v16bf load_frag_a(const __bf16* __restrict__ A, int lda, int k0) {
  const int lane = lane_id();
  const int row  = lane & 15;
  const int koff = k0 + ((lane & 16) ? 8 : 0);
  const __bf16* p = A + (size_t)row * lda + koff;   // 16B aligned
  union { v16bf v; float4 f[2]; } u;
  u.f[0] = *(const float4*)(p);        // K koff..koff+7
  u.f[1] = *(const float4*)(p + 16);   // K koff+16..koff+23
  return u.v;
}

__device__ __forceinline__ v16bf load_frag_b(const __bf16* __restrict__ B, int ldb, int k0) {
  const int lane = lane_id();
  const int col  = lane & 15;
  const int kk   = k0 + ((lane & 16) ? 8 : 0);
  const __bf16* p = B + (size_t)col * ldb + kk;     // two 16B runs: K kk..+7, kk+16..+23
  union { v16bf v; float4 f[2]; } u;
  u.f[0] = *(const float4*)(p);
  u.f[1] = *(const float4*)(p + 16);
  return u.v;
}

// C/D f32 16x16: lane<16 -> col=lane, VGPR i = row i; lane>=16 -> row 8+i.
template <bool OBF>
__device__ __forceinline__ void store_ep(float* __restrict__ C, __bf16* __restrict__ Cb,
                                         int ldc, v8f acc, float alpha, int causal,
                                         int gm0, int gn0) {
  const int lane = lane_id();
  const int col  = lane & 15;
  const int r0   = (lane & 16) ? 8 : 0;
#pragma unroll
  for (int i = 0; i < 8; ++i) {
    float v = acc[i] * alpha;
    if (causal && (gn0 + col) > (gm0 + r0 + i)) v += NEG_INF_V;
    if (OBF) Cb[(size_t)(r0 + i) * ldc + col] = (__bf16)v;
    else     C [(size_t)(r0 + i) * ldc + col] = v;
  }
}

// ---------------------------------------------------------------------------
// Unified batched GEMM:  C[z][m,n] = alpha * sum_k A[z][m,k]*B[z][n,k] (+mask)
// z decomposed zo=z/hDiv, zi=z%hDiv with independent strides (covers plain
// GEMM, per-(b,h) causal scores, and trapezoidal PV).
// 256 thr = 8 waves (4x2); wave tile 32x32 (4 WMMA accums); WG tile 128x64.
// ---------------------------------------------------------------------------
template <bool OBF>
__global__ __launch_bounds__(256)
void wmma_gemm_kernel(const __bf16* __restrict__ A, const __bf16* __restrict__ B,
                      float* __restrict__ C, __bf16* __restrict__ Cb,
                      int M, int N, int K, int lda, int ldb, int ldc,
                      long long aOuter, long long aInner,
                      long long bOuter, long long bInner,
                      long long cOuter, long long cInner,
                      int hDiv, float alpha, int causal, int trapez) {
  const int z  = blockIdx.z;
  const int zo = z / hDiv, zi = z % hDiv;
  A += zo * aOuter + zi * aInner;
  B += zo * bOuter + zi * bInner;
  if (OBF) Cb += zo * cOuter + zi * cInner;
  else     C  += zo * cOuter + zi * cInner;

  const int wave = threadIdx.x >> 5;
  const int bm = blockIdx.x * 128 + (wave >> 1) * 32;
  const int bn = blockIdx.y * 64  + (wave & 1)  * 32;
  if (bm >= M || bn >= N) return;

  v8f acc00 = {}, acc01 = {}, acc10 = {}, acc11 = {};

  // Causal: a 32x32 wave tile with min(col) > max(row) is entirely masked.
  const bool fullyMasked = causal && (bn > bm + 31);
  if (!fullyMasked) {
    // Trapezoid (PV): P[m,t]==0 for t>m, so rows bm..bm+31 only need k<=bm+31.
    const int kEnd = trapez ? min(K, bm + 32) : K;
    const __bf16* Aw = A + (size_t)bm * lda;
    const __bf16* B0 = B + (size_t)bn * ldb;
    const __bf16* B1 = B + (size_t)(bn + 16) * ldb;
    for (int k0 = 0; k0 < kEnd; k0 += 32) {
      v16bf a0 = load_frag_a(Aw,                    lda, k0);
      v16bf a1 = load_frag_a(Aw + (size_t)16 * lda, lda, k0);
      v16bf b0 = load_frag_b(B0, ldb, k0);
      v16bf b1 = load_frag_b(B1, ldb, k0);
      acc00 = __builtin_amdgcn_wmma_f32_16x16x32_bf16(false, a0, false, b0, (short)0, acc00, false, false);
      acc01 = __builtin_amdgcn_wmma_f32_16x16x32_bf16(false, a0, false, b1, (short)0, acc01, false, false);
      acc10 = __builtin_amdgcn_wmma_f32_16x16x32_bf16(false, a1, false, b0, (short)0, acc10, false, false);
      acc11 = __builtin_amdgcn_wmma_f32_16x16x32_bf16(false, a1, false, b1, (short)0, acc11, false, false);
    }
  }

  const size_t base = (size_t)bm * ldc + bn;
  float*  C00  = OBF ? nullptr : C + base;
  __bf16* Cb00 = OBF ? Cb + base : nullptr;
  store_ep<OBF>(C00, Cb00, ldc, acc00, alpha, causal, bm, bn);
  store_ep<OBF>(OBF ? C00 : C00 + 16, OBF ? Cb00 + 16 : Cb00, ldc, acc01, alpha, causal, bm, bn + 16);
  store_ep<OBF>(OBF ? C00 : C00 + (size_t)16 * ldc, OBF ? Cb00 + (size_t)16 * ldc : Cb00,
                ldc, acc10, alpha, causal, bm + 16, bn);
  store_ep<OBF>(OBF ? C00 : C00 + (size_t)16 * ldc + 16, OBF ? Cb00 + (size_t)16 * ldc + 16 : Cb00,
                ldc, acc11, alpha, causal, bm + 16, bn + 16);
}

// ---------------------------------------------------------------------------
// fp32 -> bf16 copy (4 elems/thread, vectorized)
// ---------------------------------------------------------------------------
typedef __attribute__((ext_vector_type(4))) __bf16 v4bf;
__global__ void cvt_bf16_kernel(const float* __restrict__ X, __bf16* __restrict__ Y, int total) {
  const int idx = (blockIdx.x * blockDim.x + threadIdx.x) * 4;
  if (idx >= total) return;
  float4 x = *(const float4*)(X + idx);
  v4bf y;
  y[0] = (__bf16)x.x; y[1] = (__bf16)x.y; y[2] = (__bf16)x.z; y[3] = (__bf16)x.w;
  *(v4bf*)(Y + idx) = y;
}

// ---------------------------------------------------------------------------
// RoPE: read fp32 projection, write bf16 rotated. X is (BATCH*S, heads*256);
// position = row % S.  out[j]=x[j]*c - x[j+128]*s ; out[j+128]=x[j+128]*c + x[j]*s
// ---------------------------------------------------------------------------
__global__ void rope_kernel(const float* __restrict__ Xf, __bf16* __restrict__ Xb,
                            int heads, int S, int total) {
  int idx = blockIdx.x * blockDim.x + threadIdx.x;
  if (idx >= total) return;
  const int j = idx & 127;
  const int t = idx >> 7;
  const int h = t % heads;
  const int r = t / heads;
  const int s = r % S;
  const float inv = powf(10000.0f, -(float)j * (1.0f / 128.0f));
  const float ang = (float)s * inv;
  const float c = cosf(ang), sn = sinf(ang);
  const size_t off = (size_t)r * heads * HDIM + h * HDIM + j;
  const float x1 = Xf[off], x2 = Xf[off + 128];
  Xb[off]       = (__bf16)(x1 * c - x2 * sn);
  Xb[off + 128] = (__bf16)(x2 * c + x1 * sn);
}

// ---------------------------------------------------------------------------
// Vt[b, n, t] = V[b*S + t, n]   (bf16 LDS tile transpose, 32x32, 256 threads)
// ---------------------------------------------------------------------------
__global__ __launch_bounds__(256)
void transpose_kernel(const __bf16* __restrict__ V, __bf16* __restrict__ Vt, int S) {
  __shared__ __bf16 tile[32][33];
  const int b  = blockIdx.z;
  const int t0 = blockIdx.x * 32;
  const int n0 = blockIdx.y * 32;
  const int tx = threadIdx.x & 31;
  const int ty = threadIdx.x >> 5;   // 0..7
#pragma unroll
  for (int i = 0; i < 32; i += 8)
    tile[ty + i][tx] = V[(size_t)((size_t)b * S + t0 + ty + i) * HDIM + n0 + tx];
  __syncthreads();
#pragma unroll
  for (int i = 0; i < 32; i += 8)
    Vt[((size_t)b * HDIM + n0 + ty + i) * S + t0 + tx] = tile[tx][ty + i];
}

// ---------------------------------------------------------------------------
// Row softmax: writes normalized fp32 in place (attn_weights output) AND a
// bf16 copy of P for the PV GEMM. One 256-thread WG per row (S=2048).
// ---------------------------------------------------------------------------
__global__ __launch_bounds__(256)
void softmax_kernel(float* __restrict__ P, __bf16* __restrict__ Pb, int S) {
  __shared__ float red[8];
  const size_t rowoff = (size_t)blockIdx.x * S;
  float*  p  = P  + rowoff;
  __bf16* pb = Pb + rowoff;
  const int t = threadIdx.x, lane = t & 31, wv = t >> 5;
  float v[8];
  float mx = -INFINITY;
#pragma unroll
  for (int i = 0; i < 8; ++i) { v[i] = p[t + 256 * i]; mx = fmaxf(mx, v[i]); }
#pragma unroll
  for (int off = 16; off; off >>= 1) mx = fmaxf(mx, __shfl_xor(mx, off, 32));
  if (lane == 0) red[wv] = mx;
  __syncthreads();
  mx = red[0];
#pragma unroll
  for (int i = 1; i < 8; ++i) mx = fmaxf(mx, red[i]);
  float sum = 0.0f;
#pragma unroll
  for (int i = 0; i < 8; ++i) { v[i] = __expf(v[i] - mx); sum += v[i]; }
#pragma unroll
  for (int off = 16; off; off >>= 1) sum += __shfl_xor(sum, off, 32);
  __syncthreads();
  if (lane == 0) red[wv] = sum;
  __syncthreads();
  sum = 0.0f;
#pragma unroll
  for (int i = 0; i < 8; ++i) sum += red[i];
  const float invs = 1.0f / sum;
#pragma unroll
  for (int i = 0; i < 8; ++i) {
    const float w = v[i] * invs;
    p [t + 256 * i] = w;
    pb[t + 256 * i] = (__bf16)w;
  }
}

// ---------------------------------------------------------------------------
// Launch. Inputs: 0=hidden_states 1=attention_mask(unused, analytic causal)
// 2=position_ids(unused, arange) 3=Wq 4=Wk 5=Wv 6=Wo.
// d_out = [attn_output (2*2048*2048) | attn_weights (2*8*2048*2048)] f32.
// Workspace: bf16 copies of all GEMM operands + fp32 Q/K staging, ~250 MB.
// ---------------------------------------------------------------------------
extern "C" void kernel_launch(void* const* d_in, const int* in_sizes, int n_in,
                              void* d_out, int out_size, void* d_ws, size_t ws_size,
                              hipStream_t stream) {
  (void)in_sizes; (void)n_in; (void)out_size; (void)ws_size;
  const float* H  = (const float*)d_in[0];
  const float* Wq = (const float*)d_in[3];
  const float* Wk = (const float*)d_in[4];
  const float* Wv = (const float*)d_in[5];
  const float* Wo = (const float*)d_in[6];

  float* out = (float*)d_out;
  float* AW  = out + (size_t)BATCH * S_LEN * DMODEL;   // attn_weights region

  const size_t M = (size_t)BATCH * S_LEN;              // 4096
  __bf16* Hb  = (__bf16*)d_ws;                         // M x 2048
  __bf16* Wqb = Hb  + M * DMODEL;                      // 2048 x 2048
  __bf16* Wkb = Wqb + (size_t)DMODEL * DMODEL;         // 256 x 2048
  __bf16* Wvb = Wkb + (size_t)HDIM * DMODEL;           // 256 x 2048
  __bf16* Wob = Wvb + (size_t)HDIM * DMODEL;           // 2048 x 2048
  __bf16* Qb  = Wob + (size_t)DMODEL * DMODEL;         // M x 2048 (post-RoPE)
  __bf16* Kbb = Qb  + M * DMODEL;                      // M x 256  (post-RoPE)
  __bf16* Vbb = Kbb + M * HDIM;                        // M x 256
  __bf16* Vtb = Vbb + M * HDIM;                        // B x 256 x S
  __bf16* AOb = Vtb + (size_t)BATCH * HDIM * S_LEN;    // M x 2048
  __bf16* Pb  = AOb + M * DMODEL;                      // B*H x S x S (bf16 P)
  float*  Qf  = (float*)(Pb + (size_t)BATCH * HEADS * S_LEN * S_LEN);  // M x 2048
  float*  Kf  = Qf + M * DMODEL;                       // M x 256

  dim3 blk(256);
  auto cvt = [&](const float* src, __bf16* dst, size_t n) {
    cvt_bf16_kernel<<<(unsigned)((n / 4 + 255) / 256), blk, 0, stream>>>(src, dst, (int)n);
  };

  // 0) one-time bf16 staging of H and weights
  cvt(H,  Hb,  M * DMODEL);
  cvt(Wq, Wqb, (size_t)DMODEL * DMODEL);
  cvt(Wk, Wkb, (size_t)HDIM * DMODEL);
  cvt(Wv, Wvb, (size_t)HDIM * DMODEL);
  cvt(Wo, Wob, (size_t)DMODEL * DMODEL);

  // 1) projections: Q,K in fp32 (RoPE input), V directly bf16
  wmma_gemm_kernel<false><<<dim3(M/128, DMODEL/64, 1), blk, 0, stream>>>(
      Hb, Wqb, Qf, nullptr, (int)M, DMODEL, DMODEL, DMODEL, DMODEL, DMODEL,
      0,0, 0,0, 0,0, 1, 1.0f, 0, 0);
  wmma_gemm_kernel<false><<<dim3(M/128, HDIM/64, 1), blk, 0, stream>>>(
      Hb, Wkb, Kf, nullptr, (int)M, HDIM, DMODEL, DMODEL, DMODEL, HDIM,
      0,0, 0,0, 0,0, 1, 1.0f, 0, 0);
  wmma_gemm_kernel<true><<<dim3(M/128, HDIM/64, 1), blk, 0, stream>>>(
      Hb, Wvb, nullptr, Vbb, (int)M, HDIM, DMODEL, DMODEL, DMODEL, HDIM,
      0,0, 0,0, 0,0, 1, 1.0f, 0, 0);

  // 2) RoPE: fp32 in -> bf16 out
  const int totQ = (int)(M * HEADS * (HDIM / 2));
  rope_kernel<<<(totQ + 255) / 256, blk, 0, stream>>>(Qf, Qb, HEADS, S_LEN, totQ);
  const int totK = (int)(M * 1 * (HDIM / 2));
  rope_kernel<<<(totK + 255) / 256, blk, 0, stream>>>(Kf, Kbb, 1, S_LEN, totK);

  // 3) V transpose (bf16) -> Vtb (B, HDIM, S)
  transpose_kernel<<<dim3(S_LEN/32, HDIM/32, BATCH), blk, 0, stream>>>(Vbb, Vtb, S_LEN);

  // 4) scores = (Q Kᵀ)/16 + causal -> AW fp32 (raw), z=(b*8+h)
  wmma_gemm_kernel<false><<<dim3(S_LEN/128, S_LEN/64, BATCH*HEADS), blk, 0, stream>>>(
      Qb, Kbb, AW, nullptr, S_LEN, S_LEN, HDIM, DMODEL, HDIM, S_LEN,
      (long long)S_LEN * DMODEL, (long long)HDIM,                    // A strides (b,h)
      (long long)S_LEN * HDIM,   0,                                  // B strides (shared KV)
      (long long)HEADS * S_LEN * S_LEN, (long long)S_LEN * S_LEN,    // C strides
      HEADS, 0.0625f, /*causal=*/1, 0);

  // 5) softmax rows in place -> attn_weights fp32 output + bf16 Pb
  softmax_kernel<<<BATCH * HEADS * S_LEN, blk, 0, stream>>>(AW, Pb, S_LEN);

  // 6) AO[b,:,h*256:+256] = P @ V (bf16 in, bf16 out, trapezoidal K)
  wmma_gemm_kernel<true><<<dim3(S_LEN/128, HDIM/64, BATCH*HEADS), blk, 0, stream>>>(
      Pb, Vtb, nullptr, AOb, S_LEN, HDIM, S_LEN, S_LEN, S_LEN, DMODEL,
      (long long)HEADS * S_LEN * S_LEN, (long long)S_LEN * S_LEN,
      (long long)HDIM * S_LEN, 0,
      (long long)S_LEN * DMODEL, (long long)HDIM,
      HEADS, 1.0f, 0, /*trapez=*/1);

  // 7) attn_output = AO @ Woᵀ -> fp32 d_out
  wmma_gemm_kernel<false><<<dim3(M/128, DMODEL/64, 1), blk, 0, stream>>>(
      AOb, Wob, out, nullptr, (int)M, DMODEL, DMODEL, DMODEL, DMODEL, DMODEL,
      0,0, 0,0, 0,0, 1, 1.0f, 0, 0);
}
